// PointnetFPModule_24455543783472
// MI455X (gfx1250) — matrix operand order repaired and checked by reading, not compiled, not run on hardware
//
#include <hip/hip_runtime.h>

typedef __attribute__((ext_vector_type(16))) _Float16 v16h;
typedef __attribute__((ext_vector_type(8)))  _Float16 v8h;
typedef __attribute__((ext_vector_type(8)))  float    v8f;
typedef int v4i __attribute__((vector_size(16)));

#define B_    8
#define N_    8192
#define M_    2048
#define C1_   128
#define C2_   256
#define CIN1  384    // C1_ + C2_
#define CO    256

// ---------------------------------------------------------------------------
// Async global->LDS staging helpers (guarded: async path when toolchain has
// the gfx1250 builtins, plain load+ds_store fallback otherwise).
// Builtin prototype (from hipcc diagnostic): first arg is AS1 v4i*, second is
// AS3 v4i*, then two immediate ints (offset, cpol).
// ---------------------------------------------------------------------------
#if __has_builtin(__builtin_amdgcn_global_load_async_to_lds_b128)
#define HAS_ASYNC_LDS 1
__device__ __forceinline__ void stage16(const _Float16* g, _Float16* l) {
    __builtin_amdgcn_global_load_async_to_lds_b128(
        (__attribute__((address_space(1))) v4i*)g,
        (__attribute__((address_space(3))) v4i*)l, 0, 0);
}
#if __has_builtin(__builtin_amdgcn_s_wait_asynccnt)
#define ASYNC_WAIT() __builtin_amdgcn_s_wait_asynccnt(0)
#else
#define ASYNC_WAIT() asm volatile("s_wait_asynccnt 0x0" ::: "memory")
#endif
#else
#define HAS_ASYNC_LDS 0
__device__ __forceinline__ void stage16(const _Float16* g, _Float16* l) {
    *(v8h*)l = *(const v8h*)g;          // vmem load + ds_store fallback
}
#define ASYNC_WAIT() ((void)0)
#endif

// ---------------------------------------------------------------------------
// Kernel 1: brute-force 3-NN + inverse-distance weights.
// One block = 256 query points of one batch; known xyz staged in LDS (24 KB).
// ---------------------------------------------------------------------------
__global__ void k_nn(const float* __restrict__ unknown,  // (B,n,3)
                     const float* __restrict__ known,    // (B,m,3)
                     int*   __restrict__ idx,            // (B,n,3)
                     float* __restrict__ wgt)            // (B,n,3)
{
    __shared__ float kx[M_ * 3];
    const int b = blockIdx.y;
    const float* kb = known + (size_t)b * M_ * 3;
    for (int j = threadIdx.x; j < M_ * 3; j += 256) kx[j] = kb[j];
    __syncthreads();

    const int p = blockIdx.x * 256 + threadIdx.x;      // 0..N_-1
    const float* u = unknown + ((size_t)b * N_ + p) * 3;
    const float ux = u[0], uy = u[1], uz = u[2];

    float d0 = 3.4e38f, d1 = 3.4e38f, d2 = 3.4e38f;
    int   i0 = 0, i1 = 0, i2 = 0;
    for (int j = 0; j < M_; ++j) {
        const float dx = ux - kx[3 * j + 0];
        const float dy = uy - kx[3 * j + 1];
        const float dz = uz - kx[3 * j + 2];
        const float d  = dx * dx + dy * dy + dz * dz;
        if (d < d0)      { d2 = d1; i2 = i1; d1 = d0; i1 = i0; d0 = d; i0 = j; }
        else if (d < d1) { d2 = d1; i2 = i1; d1 = d;  i1 = j; }
        else if (d < d2) { d2 = d;  i2 = j; }
    }
    const float r0 = 1.0f / (sqrtf(fmaxf(d0, 0.0f)) + 1e-8f);
    const float r1 = 1.0f / (sqrtf(fmaxf(d1, 0.0f)) + 1e-8f);
    const float r2 = 1.0f / (sqrtf(fmaxf(d2, 0.0f)) + 1e-8f);
    const float rs = 1.0f / (r0 + r1 + r2);

    const size_t o = ((size_t)b * N_ + p) * 3;
    idx[o + 0] = i0; idx[o + 1] = i1; idx[o + 2] = i2;
    wgt[o + 0] = r0 * rs; wgt[o + 1] = r1 * rs; wgt[o + 2] = r2 * rs;
}

// ---------------------------------------------------------------------------
// Kernel 2: convert W1 (256x384) and W2 (256x256) fp32 -> f16 (row-major O x K)
// ---------------------------------------------------------------------------
__global__ void k_cvtw(const float* __restrict__ W1, const float* __restrict__ W2,
                       _Float16* __restrict__ W1h, _Float16* __restrict__ W2h)
{
    const int i = blockIdx.x * 256 + threadIdx.x;
    const int n1 = CO * CIN1;            // 98304
    const int n2 = CO * C2_;             // 65536
    if (i < n1)            W1h[i]       = (_Float16)W1[i];
    else if (i < n1 + n2)  W2h[i - n1]  = (_Float16)W2[i - n1];
}

// ---------------------------------------------------------------------------
// Kernel 3: three_interpolate + concat, written as f16 in N-major layout:
//   featsT[b][n][c], c in [0,256) = interpolated known_feats, [256,384) = copy
// ---------------------------------------------------------------------------
__global__ void k_interp(const float* __restrict__ kf,   // (B,C2,m)
                         const float* __restrict__ uf,   // (B,C1,n)
                         const int*   __restrict__ idx,
                         const float* __restrict__ wgt,
                         _Float16* __restrict__ featsT)  // (B,n,384)
{
    const size_t i = (size_t)blockIdx.x * 256 + threadIdx.x;   // over B*N*384
    const int c  = (int)(i % CIN1);
    const size_t pn = i / CIN1;
    const int n  = (int)(pn % N_);
    const int b  = (int)(pn / N_);
    float v;
    if (c < C2_) {
        const size_t o = ((size_t)b * N_ + n) * 3;
        const int   j0 = idx[o + 0], j1 = idx[o + 1], j2 = idx[o + 2];
        const float w0 = wgt[o + 0], w1 = wgt[o + 1], w2 = wgt[o + 2];
        const float* f = kf + ((size_t)b * C2_ + c) * M_;
        v = f[j0] * w0 + f[j1] * w1 + f[j2] * w2;
    } else {
        v = uf[((size_t)b * C1_ + (c - C2_)) * N_ + n];
    }
    featsT[i] = (_Float16)v;
}

// ---------------------------------------------------------------------------
// Kernel 4: WMMA GEMM with async double-buffered LDS staging of B.
//   One block (8 waves) computes a 256(M) x 64(N) stripe of Y[b] for one
//   64-column group. The 64x32 f16 B tile (4 KB) is DMA'd global->LDS once
//   per K-step and shared by all 8 waves (each wave: 2 M-tiles x 4 N-tiles
//   = 8 v_wmma_f32_16x16x32_f16 per K-step).
// ---------------------------------------------------------------------------
__global__ void __launch_bounds__(256)
k_gemm(const _Float16* __restrict__ A,    // 256 x K, row-major f16
       const _Float16* __restrict__ Bm,   // B x N x K, N-major f16
       float* __restrict__ Y,             // B x 256 x N fp32
       int K)
{
    __shared__ __align__(32) _Float16 lb[2][64 * 32];   // double-buffered 4 KB

    const int tid  = threadIdx.x;
    const int wv   = tid >> 5;           // 0..7
    const int lane = tid & 31;
    const int h    = lane >> 4;          // half-wave select
    const int l    = lane & 15;

    const int nG = blockIdx.x & 127;     // 128 groups of 64 columns
    const int b  = blockIdx.x >> 7;      // batch

    // --- staging assignment: thread t copies 16 bytes of the 64x32 tile ---
    const int r  = tid >> 2;             // LDS row (= local n), 0..63
    const int cc = (tid & 3) * 8;        // half-offset within 32-half row
    const _Float16* gsrc = Bm + ((size_t)b * N_ + (size_t)nG * 64 + r) * K + cc;
    _Float16* ldst0 = &lb[0][r * 32 + cc];
    _Float16* ldst1 = &lb[1][r * 32 + cc];

    // --- A rows for this wave's two M-tiles (rows 32*wv .. 32*wv+31) ---
    const _Float16* Arow0 = A + (size_t)(wv * 32 +      l) * K;
    const _Float16* Arow1 = A + (size_t)(wv * 32 + 16 + l) * K;

    v8f acc[2][4];
    #pragma unroll
    for (int mi = 0; mi < 2; ++mi)
        #pragma unroll
        for (int ni = 0; ni < 4; ++ni) acc[mi][ni] = (v8f){};

    stage16(gsrc, ldst0);                // prologue: K-step 0 into buffer 0

    for (int k0 = 0, it = 0; k0 < K; k0 += 32, ++it) {
        const int cur = it & 1;
        ASYNC_WAIT();                    // this wave's staged tile landed
        __syncthreads();                 // all waves' tiles landed; prev reads done
        if (k0 + 32 < K)                 // launch DMA for next K-step
            stage16(gsrc + k0 + 32, cur ? ldst0 : ldst1);

        // A fragments (ISA 16-bit A 16x32 layout: two 8-half runs per lane)
        v16h a0 = __builtin_shufflevector(
            *(const v8h*)(Arow0 + k0 + 8 * h), *(const v8h*)(Arow0 + k0 + 16 + 8 * h),
            0, 1, 2, 3, 4, 5, 6, 7, 8, 9, 10, 11, 12, 13, 14, 15);
        v16h a1 = __builtin_shufflevector(
            *(const v8h*)(Arow1 + k0 + 8 * h), *(const v8h*)(Arow1 + k0 + 16 + 8 * h),
            0, 1, 2, 3, 4, 5, 6, 7, 8, 9, 10, 11, 12, 13, 14, 15);
        if (k0 + 64 < K) __builtin_prefetch((const void*)(Arow0 + k0 + 64), 0, 1);

        // B fragments from LDS (two 8-half runs per lane, same k order as A)
        #pragma unroll
        for (int ni = 0; ni < 4; ++ni) {
            v16h bt = __builtin_shufflevector(
                *(const v8h*)&lb[cur][(ni * 16 + l) * 32 + 8 * h],
                *(const v8h*)&lb[cur][(ni * 16 + l) * 32 + 16 + 8 * h],
                0, 1, 2, 3, 4, 5, 6, 7, 8, 9, 10, 11, 12, 13, 14, 15);
            acc[0][ni] = __builtin_amdgcn_wmma_f32_16x16x32_f16(
                false, a0, false, bt, (short)0, acc[0][ni], false, false);
            acc[1][ni] = __builtin_amdgcn_wmma_f32_16x16x32_f16(
                false, a1, false, bt, (short)0, acc[1][ni], false, false);
        }
        __syncthreads();                 // reads of lb[cur] done before reuse
    }

    // D layout: vgpr j, lanes 0-15 -> M=j, lanes 16-31 -> M=j+8; N = lane&15
    #pragma unroll
    for (int mi = 0; mi < 2; ++mi) {
        float* yb = Y + ((size_t)b * CO + (size_t)(wv * 32 + mi * 16 + 8 * h)) * N_
                      + (size_t)nG * 64 + l;
        #pragma unroll
        for (int j = 0; j < 8; ++j) {
            yb[(size_t)j * N_ +  0] = acc[mi][0][j];
            yb[(size_t)j * N_ + 16] = acc[mi][1][j];
            yb[(size_t)j * N_ + 32] = acc[mi][2][j];
            yb[(size_t)j * N_ + 48] = acc[mi][3][j];
        }
    }
}

// ---------------------------------------------------------------------------
// Kernel 5: per-channel batch stats over (B, n) -> fold gamma/beta into
// scale/shift: y_norm = y*scale + shift. One block per channel.
// ---------------------------------------------------------------------------
__global__ void k_stats(const float* __restrict__ Y,
                        const float* __restrict__ gamma,
                        const float* __restrict__ beta,
                        float* __restrict__ scale, float* __restrict__ shift)
{
    __shared__ float ssum[256], ssq[256];
    const int c = blockIdx.x;
    float s = 0.0f, q = 0.0f;
    for (int b = 0; b < B_; ++b) {
        const float* row = Y + ((size_t)b * CO + c) * N_;
        for (int n = threadIdx.x; n < N_; n += 256) {
            const float v = row[n];
            s += v; q += v * v;
        }
    }
    ssum[threadIdx.x] = s; ssq[threadIdx.x] = q;
    __syncthreads();
    for (int st = 128; st > 0; st >>= 1) {
        if (threadIdx.x < st) {
            ssum[threadIdx.x] += ssum[threadIdx.x + st];
            ssq[threadIdx.x]  += ssq[threadIdx.x + st];
        }
        __syncthreads();
    }
    if (threadIdx.x == 0) {
        const float inv  = 1.0f / (float)(B_ * N_);
        const float mean = ssum[0] * inv;
        const float var  = ssq[0] * inv - mean * mean;
        const float rstd = rsqrtf(var + 1e-5f);
        const float sc   = gamma[c] * rstd;
        scale[c] = sc;
        shift[c] = beta[c] - mean * sc;
    }
}

// ---------------------------------------------------------------------------
// Kernel 6: BN+ReLU, write f16 N-major (B, n, 256) for the next GEMM.
// ---------------------------------------------------------------------------
__global__ void k_bnrelu_h(const float* __restrict__ Y,
                           const float* __restrict__ scale,
                           const float* __restrict__ shift,
                           _Float16* __restrict__ xT)   // (B, n, 256)
{
    const size_t i = (size_t)blockIdx.x * 256 + threadIdx.x;  // B*N*256, c fastest
    const int c = (int)(i & 255);
    const size_t pn = i >> 8;
    const int n = (int)(pn % N_);
    const int b = (int)(pn / N_);
    float v = Y[((size_t)b * CO + c) * N_ + n] * scale[c] + shift[c];
    xT[i] = (_Float16)fmaxf(v, 0.0f);
}

// ---------------------------------------------------------------------------
// Kernel 7: BN+ReLU, final fp32 output in (B, 256, n) layout.
// ---------------------------------------------------------------------------
__global__ void k_bnrelu_out(const float* __restrict__ Y,
                             const float* __restrict__ scale,
                             const float* __restrict__ shift,
                             float* __restrict__ out)
{
    const size_t i = (size_t)blockIdx.x * 256 + threadIdx.x;  // B*256*N, n fastest
    const int c = (int)((i / N_) % CO);
    out[i] = fmaxf(Y[i] * scale[c] + shift[c], 0.0f);
}

// ---------------------------------------------------------------------------
extern "C" void kernel_launch(void* const* d_in, const int* in_sizes, int n_in,
                              void* d_out, int out_size, void* d_ws, size_t ws_size,
                              hipStream_t stream)
{
    const float* unknown = (const float*)d_in[0];
    const float* known   = (const float*)d_in[1];
    const float* uf      = (const float*)d_in[2];
    const float* kf      = (const float*)d_in[3];
    const float* W1      = (const float*)d_in[4];
    const float* g1      = (const float*)d_in[5];
    const float* b1      = (const float*)d_in[6];
    const float* W2      = (const float*)d_in[7];
    const float* g2      = (const float*)d_in[8];
    const float* b2      = (const float*)d_in[9];
    float* out = (float*)d_out;

    char* ws = (char*)d_ws;
    size_t off = 0;
    auto alloc = [&](size_t bytes) -> void* {
        void* p = ws + off;
        off += (bytes + 255) & ~(size_t)255;
        return p;
    };
    int*      idx    = (int*)     alloc((size_t)B_ * N_ * 3 * sizeof(int));
    float*    wgt    = (float*)   alloc((size_t)B_ * N_ * 3 * sizeof(float));
    _Float16* featsT = (_Float16*)alloc((size_t)B_ * N_ * CIN1 * sizeof(_Float16));
    _Float16* W1h    = (_Float16*)alloc((size_t)CO * CIN1 * sizeof(_Float16));
    _Float16* W2h    = (_Float16*)alloc((size_t)CO * C2_ * sizeof(_Float16));
    float*    y      = (float*)   alloc((size_t)B_ * CO * N_ * sizeof(float));
    _Float16* x1T    = (_Float16*)alloc((size_t)B_ * N_ * CO * sizeof(_Float16));
    float*    scale1 = (float*)   alloc(CO * sizeof(float));
    float*    shift1 = (float*)   alloc(CO * sizeof(float));
    float*    scale2 = (float*)   alloc(CO * sizeof(float));
    float*    shift2 = (float*)   alloc(CO * sizeof(float));

    // 1) 3-NN + weights
    k_nn<<<dim3(N_ / 256, B_), 256, 0, stream>>>(unknown, known, idx, wgt);
    // 2) weights fp32 -> f16
    k_cvtw<<<(CO * CIN1 + CO * C2_ + 255) / 256, 256, 0, stream>>>(W1, W2, W1h, W2h);
    // 3) interpolate + concat, f16 N-major
    k_interp<<<(int)(((size_t)B_ * N_ * CIN1) / 256), 256, 0, stream>>>(kf, uf, idx, wgt, featsT);
    // 4) conv1 GEMM (K = 384): 8 batches x 128 column-groups = 1024 blocks
    k_gemm<<<B_ * 128, 256, 0, stream>>>(W1h, featsT, y, CIN1);
    // 5) BN stats 1
    k_stats<<<CO, 256, 0, stream>>>(y, g1, b1, scale1, shift1);
    // 6) BN+ReLU -> f16 N-major for conv2
    k_bnrelu_h<<<(int)(((size_t)B_ * N_ * CO) / 256), 256, 0, stream>>>(y, scale1, shift1, x1T);
    // 7) conv2 GEMM (K = 256)
    k_gemm<<<B_ * 128, 256, 0, stream>>>(W2h, x1T, y, C2_);
    // 8) BN stats 2
    k_stats<<<CO, 256, 0, stream>>>(y, g2, b2, scale2, shift2);
    // 9) BN+ReLU -> final fp32 output
    k_bnrelu_out<<<(int)(((size_t)B_ * CO * N_) / 256), 256, 0, stream>>>(y, scale2, shift2, out);
}
